// CoAttentionModule_19791209300761
// MI455X (gfx1250) — matrix operand order repaired and checked by reading, not compiled, not run on hardware
//
#include <hip/hip_runtime.h>
#include <stdint.h>

typedef __attribute__((ext_vector_type(16))) _Float16 v16h;
typedef __attribute__((ext_vector_type(8)))  _Float16 v8h;
typedef __attribute__((ext_vector_type(8)))  float    v8f;
typedef int v4i_ __attribute__((vector_size(16)));

constexpr int B_  = 4;
constexpr int C_  = 2048;   // input channels
constexpr int NN_ = 2304;   // H*W = 48*48
constexpr int D_  = 256;    // hidden dim
constexpr int OC_ = 4096;   // output channels per side (2*C)

constexpr int LDSPITCH = 40;      // 32 k-halfs + 8 pad (bank destagger)
constexpr int LDSTILE  = 64 * LDSPITCH;

// Async global->LDS staging (CDNA5 GLOBAL_LOAD_ASYNC_TO_LDS_B128, ASYNCcnt)
#if __has_builtin(__builtin_amdgcn_global_load_async_to_lds_b128)
#define ASYNC_STAGE 1
#endif

// ---------------------------------------------------------------------------
// WMMA fragment helpers (wave32, v_wmma_f32_16x16x32_f16 layouts, ISA 7.12.2)
//   A (16x32 f16): lane L -> row m = L&15; K chunks [8h,8h+8) and [16+8h,+8)
//   B (32x16 f16): lane L -> col  = L&15; K chunks [8h,8h+8) and [16+8h,+8)
//   C/D (16x16 f32): lane L -> col = L&15; vgpr r -> row m = r + 8h
// ---------------------------------------------------------------------------
__device__ __forceinline__ v16h load_frag_A(const _Float16* __restrict__ base, int ld) {
  const int lane = threadIdx.x & 31;
  const int m    = lane & 15;
  const int half = lane >> 4;
  const _Float16* p = base + m * ld;
  v8h c0 = *reinterpret_cast<const v8h*>(p + 8 * half);
  v8h c1 = *reinterpret_cast<const v8h*>(p + 16 + 8 * half);
  v16h a;
#pragma unroll
  for (int i = 0; i < 8; ++i) { a[i] = c0[i]; a[8 + i] = c1[i]; }
  return a;
}

// B fragment from the LDS-staged tile (64 cols x 32 k, pitch LDSPITCH)
__device__ __forceinline__ v16h frag_B_lds(const _Float16* buf, int colBase) {
  const int lane = threadIdx.x & 31;
  const int col  = lane & 15;
  const int half = lane >> 4;
  const _Float16* p = buf + (colBase + col) * LDSPITCH + 8 * half;
  v8h c0 = *reinterpret_cast<const v8h*>(p);
  v8h c1 = *reinterpret_cast<const v8h*>(p + 16);
  v16h b;
#pragma unroll
  for (int i = 0; i < 8; ++i) { b[i] = c0[i]; b[8 + i] = c1[i]; }
  return b;
}

__device__ __forceinline__ v8f wmma_f16(v16h a, v16h b, v8f c) {
  return __builtin_amdgcn_wmma_f32_16x16x32_f16(false, a, false, b, (short)0, c, false, false);
}

// Cooperative stage: 256 threads copy a 64x32 f16 B^T tile (rows = out column,
// contiguous K) from global into LDS. Thread t: row = t/4, 16B chunk = t%4.
__device__ __forceinline__ void stage_B(const _Float16* __restrict__ Bt, int ldb,
                                        int k0, int K, _Float16* buf) {
  const int t   = threadIdx.x;
  const int row = t >> 2;
  const int ch  = (t & 3) * 8;
  const _Float16* src = Bt + (long)row * ldb + k0 + ch;
  _Float16*       dst = buf + row * LDSPITCH + ch;
#if defined(ASYNC_STAGE)
  // memory -> LDS directly, no VGPR round trip; tracked by ASYNCcnt.
  // Signature (from probe): (int4 AS1*, int4 AS3*, imm offset, imm cpol).
  __builtin_amdgcn_global_load_async_to_lds_b128(
      (__attribute__((address_space(1))) v4i_*)(uintptr_t)src,
      (__attribute__((address_space(3))) v4i_*)(unsigned int)(uintptr_t)dst,
      /*offset=*/0, /*cpol=*/0);
#else
  *reinterpret_cast<v8h*>(dst) = *reinterpret_cast<const v8h*>(src);
#endif
  if (k0 + 64 < K)  // confirmed lowering: global_prefetch_b8
    __builtin_prefetch(src + 64, 0, 3);
}

// Order async LDS writes before the workgroup barrier.
__device__ __forceinline__ void wait_stage() {
#if defined(ASYNC_STAGE)
#if __has_builtin(__builtin_amdgcn_s_wait_asynccnt)
  __builtin_amdgcn_s_wait_asynccnt(0);
#else
  asm volatile("s_wait_asynccnt 0" ::: "memory");
#endif
#endif
}

// ---------------------------------------------------------------------------
// Cooperative pipelined mainloop: block = 8 waves, macro-tile 256(M) x 64(N).
// Wave w owns rows [w*32, w*32+32). Per K-step: async-stage next B tile +
// prefetch next A frags, then 8 WMMAs on current data, one barrier.
// ---------------------------------------------------------------------------
__device__ __forceinline__ void coop_gemm(const _Float16* __restrict__ A, int lda,
                                          const _Float16* __restrict__ Bt, int ldb,
                                          int K, _Float16* lds0, _Float16* lds1,
                                          v8f (&acc)[2][4]) {
  stage_B(Bt, ldb, 0, K, lds0);
  v16h aC0 = load_frag_A(A, lda);
  v16h aC1 = load_frag_A(A + 16 * lda, lda);
  wait_stage();
  __syncthreads();

  for (int k0 = 0, st = 0; k0 < K; k0 += 32, st ^= 1) {
    _Float16* cur = st ? lds1 : lds0;
    _Float16* nxt = st ? lds0 : lds1;
    v16h aN0 = {}, aN1 = {};
    if (k0 + 32 < K) {
      stage_B(Bt, ldb, k0 + 32, K, nxt);          // global->LDS for next step
      aN0 = load_frag_A(A + k0 + 32, lda);        // next A frags (registers)
      aN1 = load_frag_A(A + 16 * lda + k0 + 32, lda);
    }
#pragma unroll
    for (int j = 0; j < 4; ++j) {
      v16h bf = frag_B_lds(cur, 16 * j);
      acc[0][j] = wmma_f16(aC0, bf, acc[0][j]);
      acc[1][j] = wmma_f16(aC1, bf, acc[1][j]);
    }
    aC0 = aN0; aC1 = aN1;
    wait_stage();
    __syncthreads();
  }
}

// ---------------------------------------------------------------------------
// 1) Elementwise casts
// ---------------------------------------------------------------------------
__global__ void cast_features_kernel(const float* __restrict__ left,
                                     const float* __restrict__ right,
                                     _Float16* __restrict__ vc) {
  const long half   = (long)B_ * C_ * NN_;
  const long total  = 2 * half;
  const long stride = (long)gridDim.x * blockDim.x;
  for (long i = (long)blockIdx.x * blockDim.x + threadIdx.x; i < total; i += stride) {
    float v = (i < half) ? left[i] : right[i - half];
    vc[i] = (_Float16)v;
  }
}

__global__ void cast_weights_kernel(const float* __restrict__ wq,
                                    const float* __restrict__ wr,
                                    _Float16* __restrict__ w16) {
  const int half = D_ * C_;
  int i = blockIdx.x * blockDim.x + threadIdx.x;
  if (i < 2 * half) {
    float v = (i < half) ? wq[i] : wr[i - half];
    w16[i] = (_Float16)v;
  }
}

// ---------------------------------------------------------------------------
// 2) Transpose-cast: Vt16[s][b][n][c] = (f16) feat_s[b][c][n]
// ---------------------------------------------------------------------------
__global__ void transpose_cast_kernel(const float* __restrict__ left,
                                      const float* __restrict__ right,
                                      _Float16* __restrict__ vt) {
  __shared__ _Float16 tile[32][33];
  const int sb = blockIdx.z;
  const int s  = sb >> 2;
  const int b  = sb & 3;
  const float* src = (s ? right : left) + (long)b * C_ * NN_;
  _Float16*    dst = vt + (long)sb * NN_ * C_;
  const int n0 = blockIdx.x * 32;
  const int c0 = blockIdx.y * 32;
  const int tx = threadIdx.x, ty = threadIdx.y;
#pragma unroll
  for (int i = 0; i < 32; i += 8)
    tile[ty + i][tx] = (_Float16)src[(long)(c0 + ty + i) * NN_ + n0 + tx];
  __syncthreads();
#pragma unroll
  for (int i = 0; i < 32; i += 8)
    dst[(long)(n0 + ty + i) * C_ + c0 + tx] = tile[tx][ty + i];
}

// ---------------------------------------------------------------------------
// 3) Projection GEMM: out[b][n][d] = W[d][:] . Vt[b][n][:] + bias[d]  (f16 out)
//    grid.x = B * (NN/64); M = D = 256 rows = one full block macro-tile.
// ---------------------------------------------------------------------------
__global__ void proj_kernel(const _Float16* __restrict__ W,    // [D][C]
                            const _Float16* __restrict__ Vt,   // [B][NN][C]
                            const float*    __restrict__ bias,
                            _Float16*       __restrict__ out)  // [B][NN][D]
{
  __shared__ _Float16 lds[2][LDSTILE];
  const int NT = NN_ / 64;                 // 36
  const int b  = blockIdx.x / NT;
  const int n0 = (blockIdx.x % NT) * 64;
  const int wave = threadIdx.x >> 5;

  const _Float16* A  = W + (wave * 32) * C_;
  const _Float16* Bt = Vt + (long)b * NN_ * C_ + (long)n0 * C_;

  const v8f z = {0.f, 0.f, 0.f, 0.f, 0.f, 0.f, 0.f, 0.f};
  v8f acc[2][4] = {{z, z, z, z}, {z, z, z, z}};
  coop_gemm(A, C_, Bt, C_, C_, lds[0], lds[1], acc);

  const int lane = threadIdx.x & 31;
  const int col  = lane & 15;
  const int half = lane >> 4;
#pragma unroll
  for (int mf = 0; mf < 2; ++mf) {
    const int dbase = wave * 32 + mf * 16;
    float bv[8];
#pragma unroll
    for (int rr = 0; rr < 8; ++rr) bv[rr] = bias[dbase + 8 * half + rr];
#pragma unroll
    for (int j = 0; j < 4; ++j) {
      v8h o;
#pragma unroll
      for (int rr = 0; rr < 8; ++rr) o[rr] = (_Float16)(acc[mf][j][rr] + bv[rr]);
      *reinterpret_cast<v8h*>(out + (long)b * NN_ * D_ +
                              (long)(n0 + 16 * j + col) * D_ + dbase + 8 * half) = o;
    }
  }
}

// ---------------------------------------------------------------------------
// 4) Scores GEMM: S[b][n][m] = Q[b][n][:] . K[b][m][:]   (f32 out)
//    grid.x = B * (NN/256) * (NN/64)
// ---------------------------------------------------------------------------
__global__ void scores_kernel(const _Float16* __restrict__ Q,  // [B][NN][D]
                              const _Float16* __restrict__ Kx, // [B][NN][D]
                              float* __restrict__ S)           // [B][NN][NN]
{
  __shared__ _Float16 lds[2][LDSTILE];
  const int MB = NN_ / 256, NT = NN_ / 64;       // 9, 36
  const int b  = blockIdx.x / (MB * NT);
  const int r  = blockIdx.x % (MB * NT);
  const int nB = (r / NT) * 256;
  const int m0 = (r % NT) * 64;
  const int wave   = threadIdx.x >> 5;
  const int n_base = nB + wave * 32;

  const _Float16* A  = Q  + (long)b * NN_ * D_ + (long)n_base * D_;
  const _Float16* Bt = Kx + (long)b * NN_ * D_ + (long)m0 * D_;

  const v8f z = {0.f, 0.f, 0.f, 0.f, 0.f, 0.f, 0.f, 0.f};
  v8f acc[2][4] = {{z, z, z, z}, {z, z, z, z}};
  coop_gemm(A, D_, Bt, D_, D_, lds[0], lds[1], acc);

  const int lane = threadIdx.x & 31;
  const int col  = lane & 15;
  const int half = lane >> 4;
  float* Sb = S + (long)b * NN_ * NN_;
#pragma unroll
  for (int mf = 0; mf < 2; ++mf)
#pragma unroll
    for (int j = 0; j < 4; ++j)
#pragma unroll
      for (int rr = 0; rr < 8; ++rr)
        Sb[(long)(n_base + mf * 16 + rr + 8 * half) * NN_ + m0 + 16 * j + col] =
            acc[mf][j][rr];
}

// ---------------------------------------------------------------------------
// 5) Row softmax: P16[row][:] = softmax(S[row][:]); 2304 = 9*256
// ---------------------------------------------------------------------------
__global__ void softmax_kernel(const float* __restrict__ S, _Float16* __restrict__ P) {
  const long row = blockIdx.x;
  const float* src = S + row * NN_;
  _Float16*    dst = P + row * NN_;
  __shared__ float red[256];
  const int tid = threadIdx.x;

  float vals[9];
  float m = -3.0e38f;
#pragma unroll
  for (int i = 0; i < 9; ++i) { vals[i] = src[tid + i * 256]; m = fmaxf(m, vals[i]); }
  red[tid] = m; __syncthreads();
  for (int s = 128; s > 0; s >>= 1) {
    if (tid < s) red[tid] = fmaxf(red[tid], red[tid + s]);
    __syncthreads();
  }
  m = red[0]; __syncthreads();

  float sum = 0.f;
#pragma unroll
  for (int i = 0; i < 9; ++i) { vals[i] = __expf(vals[i] - m); sum += vals[i]; }
  red[tid] = sum; __syncthreads();
  for (int s = 128; s > 0; s >>= 1) {
    if (tid < s) red[tid] += red[tid + s];
    __syncthreads();
  }
  const float inv = 1.0f / red[0];
#pragma unroll
  for (int i = 0; i < 9; ++i) dst[tid + i * 256] = (_Float16)(vals[i] * inv);
}

// ---------------------------------------------------------------------------
// 6) AV GEMM: out[b][2048+c][n] = sum_m P[b][n][m] * Vc[b][c][m]
//    grid.x = B * (NN/256) * (C/64)
// ---------------------------------------------------------------------------
__global__ void av_kernel(const _Float16* __restrict__ P,   // [B][NN][NN]
                          const _Float16* __restrict__ Vc,  // [B][C][NN]
                          float* __restrict__ out)          // [B][OC][NN] side base
{
  __shared__ _Float16 lds[2][LDSTILE];
  const int MB = NN_ / 256, NT = C_ / 64;        // 9, 32
  const int b  = blockIdx.x / (MB * NT);
  const int r  = blockIdx.x % (MB * NT);
  const int nB = (r / NT) * 256;
  const int c0 = (r % NT) * 64;
  const int wave   = threadIdx.x >> 5;
  const int n_base = nB + wave * 32;

  const _Float16* A  = P  + (long)b * NN_ * NN_ + (long)n_base * NN_;
  const _Float16* Bt = Vc + (long)b * C_  * NN_ + (long)c0 * NN_;

  const v8f z = {0.f, 0.f, 0.f, 0.f, 0.f, 0.f, 0.f, 0.f};
  v8f acc[2][4] = {{z, z, z, z}, {z, z, z, z}};
  coop_gemm(A, NN_, Bt, NN_, NN_, lds[0], lds[1], acc);

  const int lane = threadIdx.x & 31;
  const int col  = lane & 15;
  const int half = lane >> 4;
#pragma unroll
  for (int mf = 0; mf < 2; ++mf)
#pragma unroll
    for (int j = 0; j < 4; ++j) {
      float* op = out + ((long)b * OC_ + 2048 + c0 + 16 * j + col) * NN_ +
                  n_base + mf * 16 + 8 * half;
#pragma unroll
      for (int rr = 0; rr < 8; ++rr) op[rr] = acc[mf][j][rr];
    }
}

// ---------------------------------------------------------------------------
extern "C" void kernel_launch(void* const* d_in, const int* in_sizes, int n_in,
                              void* d_out, int out_size, void* d_ws, size_t ws_size,
                              hipStream_t stream) {
  (void)in_sizes; (void)n_in; (void)out_size; (void)ws_size;
  const float* left  = (const float*)d_in[0];
  const float* right = (const float*)d_in[1];
  const float* wq    = (const float*)d_in[2];
  const float* bq    = (const float*)d_in[3];
  const float* wr    = (const float*)d_in[4];
  const float* br    = (const float*)d_in[5];
  float* out = (float*)d_out;

  // workspace carve-up (256B aligned regions)
  char* ws = (char*)d_ws;
  size_t off = 0;
  auto carve = [&](size_t bytes) -> void* {
    void* p = ws + off;
    off += (bytes + 255) & ~(size_t)255;
    return p;
  };
  _Float16* Vt  = (_Float16*)carve((size_t)2 * B_ * NN_ * C_ * 2); // [s][b][n][c]
  _Float16* Vc  = (_Float16*)carve((size_t)2 * B_ * C_ * NN_ * 2); // [s][b][c][n]
  _Float16* W16 = (_Float16*)carve((size_t)2 * D_ * C_ * 2);       // [wq | wr], [d][c]
  _Float16* Q16 = (_Float16*)carve((size_t)2 * B_ * NN_ * D_ * 2); // [s][b][n][d]
  _Float16* K16 = (_Float16*)carve((size_t)2 * B_ * NN_ * D_ * 2);
  float*    S   = (float*)   carve((size_t)B_ * NN_ * NN_ * 4);    // reused per dir
  _Float16* P16 = (_Float16*)carve((size_t)B_ * NN_ * NN_ * 2);

  const long featPerB = (long)C_ * NN_;        // elems per (b) per side
  const long sideOut  = (long)B_ * OC_ * NN_;  // elems per output side

  // 1) casts
  cast_features_kernel<<<8192, 256, 0, stream>>>(left, right, Vc);
  cast_weights_kernel<<<(2 * D_ * C_) / 256, 256, 0, stream>>>(wq, wr, W16);

  // 2) transpose-cast
  {
    dim3 g(NN_ / 32, C_ / 32, 2 * B_), blk(32, 8);
    transpose_cast_kernel<<<g, blk, 0, stream>>>(left, right, Vt);
  }

  // 3) passthrough halves: out[side][b][0:2048][:] = feat_side[b]
  for (int s = 0; s < 2; ++s) {
    const float* src = s ? right : left;
    for (int b = 0; b < B_; ++b) {
      (void)hipMemcpyAsync(out + s * sideOut + (long)b * OC_ * NN_,
                           src + (long)b * featPerB,
                           (size_t)featPerB * sizeof(float),
                           hipMemcpyDeviceToDevice, stream);
    }
  }

  // 4) projections: Q[s] = Vt[s] . wq^T + bq ; K[s] = Vt[s] . wr^T + br
  {
    const int blocks = B_ * (NN_ / 64);              // 144
    for (int s = 0; s < 2; ++s) {
      const _Float16* vts = Vt + (long)s * B_ * NN_ * C_;
      proj_kernel<<<blocks, 256, 0, stream>>>(W16,           vts, bq,
                                              Q16 + (long)s * B_ * NN_ * D_);
      proj_kernel<<<blocks, 256, 0, stream>>>(W16 + D_ * C_, vts, br,
                                              K16 + (long)s * B_ * NN_ * D_);
    }
  }

  // 5) per-direction attention
  for (int dir = 0; dir < 2; ++dir) {
    const int sq = dir, sk = 1 - dir;
    scores_kernel<<<B_ * (NN_ / 256) * (NN_ / 64), 256, 0, stream>>>(
        Q16 + (long)sq * B_ * NN_ * D_,
        K16 + (long)sk * B_ * NN_ * D_, S);
    softmax_kernel<<<B_ * NN_, 256, 0, stream>>>(S, P16);
    av_kernel<<<B_ * (NN_ / 256) * (C_ / 64), 256, 0, stream>>>(
        P16, Vc + (long)sk * B_ * C_ * NN_, out + dir * sideOut);
  }
}
